// MultiHeadAttention_83958020702256
// MI455X (gfx1250) — compile-verified, matrix operand order checked by reading, not run on hardware
//
#include <hip/hip_runtime.h>

// ---------------------------------------------------------------------------
// Types for CDNA5 WMMA (gfx1250): v_wmma_f32_16x16x32_bf16
// ---------------------------------------------------------------------------
typedef __attribute__((ext_vector_type(16))) __bf16 v16bf;
typedef __attribute__((ext_vector_type(8)))  __bf16 v8bf;
typedef __attribute__((ext_vector_type(4)))  __bf16 v4bf;
typedef __attribute__((ext_vector_type(8)))  float  v8f;
typedef __attribute__((ext_vector_type(4)))  float  v4f;

#define WMMA_BF16(A, B, C) \
  __builtin_amdgcn_wmma_f32_16x16x32_bf16(false, (A), false, (B), (short)0, (C), false, false)

__device__ __forceinline__ int lane_id() {
  // wave32: popcount of exec-low bits below this lane
  return __builtin_amdgcn_mbcnt_lo(~0u, 0u);
}

__device__ __forceinline__ float shfl_xor_f(float v, int mask) {
  int l = lane_id();
  return __int_as_float(__builtin_amdgcn_ds_bpermute(((l ^ mask) << 2), __float_as_int(v)));
}

// Native fp32 -> bf16 (v_cvt_*_bf16), packed where possible.
__device__ __forceinline__ v4bf cvt4(v4f f) {
  return __builtin_convertvector(f, v4bf);
}

// Build a 16x32 bf16 WMMA fragment from two contiguous 8-element LDS chunks.
// ISA layout: lanes 0-15 row=lane&15 hold K {0..7} (VGPR0-3) and {16..23} (VGPR4-7);
// lanes 16-31 hold K {8..15} and {24..31}.  lo = base + (lane>>4)*8, hi = lo + 16.
__device__ __forceinline__ v16bf frag16(const __bf16* lo, const __bf16* hi) {
  union { v16bf v; v8bf h[2]; } u;
  u.h[0] = *(const v8bf*)lo;
  u.h[1] = *(const v8bf*)hi;
  return u.v;
}

__device__ __forceinline__ v8f zero8() {
  v8f z;
#pragma unroll
  for (int i = 0; i < 8; ++i) z[i] = 0.0f;
  return z;
}

// ---------------------------------------------------------------------------
// Problem constants
// ---------------------------------------------------------------------------
#define EMBED 1024
#define HEADS 16
#define HDIM  64
#define BATCH 4
#define SEQ   2048
#define MTOT  (BATCH * SEQ)   // 8192 rows

// GEMM tiling
#define BM 128
#define BN 128
#define BK 32
#define LDA 40   // padded LDS stride (bf16 elems), 80B = 16B-aligned, bank-spread

// ---------------------------------------------------------------------------
// Kernel 1: QKV projection.  qkv = x @ Wqkv^T + b, scattered to Q/K/V [B,H,T,Dh] bf16.
// Q is pre-scaled by 1/sqrt(Dh).
// ---------------------------------------------------------------------------
__global__ __launch_bounds__(256) void qkv_gemm_kernel(
    const float* __restrict__ X,     // [8192,1024]
    const float* __restrict__ W,     // [3072,1024] row-major (output col j <-> row j)
    const float* __restrict__ bias,  // [3072]
    __bf16* __restrict__ Qb, __bf16* __restrict__ Kb, __bf16* __restrict__ Vb)
{
  __shared__ __bf16 As[BM * LDA];
  __shared__ __bf16 Bs[BN * LDA];

  const int tid  = threadIdx.x;
  const int lane = tid & 31;
  const int wave = tid >> 5;      // 0..7
  const int wm   = wave >> 2;     // 0..1
  const int wn   = wave & 3;      // 0..3
  const int hi   = lane >> 4;     // 0/1
  const int l    = lane & 15;

  const int m0 = blockIdx.y * BM;
  const int n0 = blockIdx.x * BN;

  v8f acc[4][2];
#pragma unroll
  for (int mf = 0; mf < 4; ++mf)
#pragma unroll
    for (int nf = 0; nf < 2; ++nf) acc[mf][nf] = zero8();

  for (int k0 = 0; k0 < EMBED; k0 += BK) {
    // ---- global -> LDS (fp32 -> bf16 via packed cvt) : A tile 128x32
#pragma unroll
    for (int i = 0; i < 4; ++i) {
      int idx = tid + i * 256;          // float4 index 0..1023
      int row = idx >> 3;               // 8 float4 per row
      int c4  = idx & 7;
      v4f f = *(const v4f*)(X + (size_t)(m0 + row) * EMBED + k0 + c4 * 4);
      *(v4bf*)&As[row * LDA + c4 * 4] = cvt4(f);
    }
    // ---- B tile 128x32 (rows of W)
#pragma unroll
    for (int i = 0; i < 4; ++i) {
      int idx = tid + i * 256;
      int row = idx >> 3;
      int c4  = idx & 7;
      v4f f = *(const v4f*)(W + (size_t)(n0 + row) * EMBED + k0 + c4 * 4);
      *(v4bf*)&Bs[row * LDA + c4 * 4] = cvt4(f);
    }
    if (k0 + BK < EMBED)  // gfx1250 global_prefetch_b8 of next A tile
      __builtin_prefetch(X + (size_t)(m0 + (tid >> 1)) * EMBED + k0 + BK, 0, 1);
    __syncthreads();

    // ---- fragments + WMMA
    v16bf bf[2];
#pragma unroll
    for (int nf = 0; nf < 2; ++nf) {
      const __bf16* base = &Bs[(wn * 32 + nf * 16 + l) * LDA];
      bf[nf] = frag16(base + hi * 8, base + 16 + hi * 8);
    }
#pragma unroll
    for (int mf = 0; mf < 4; ++mf) {
      const __bf16* abase = &As[(wm * 64 + mf * 16 + l) * LDA];
      v16bf af = frag16(abase + hi * 8, abase + 16 + hi * 8);
#pragma unroll
      for (int nf = 0; nf < 2; ++nf)
        acc[mf][nf] = WMMA_BF16(af, bf[nf], acc[mf][nf]);
    }
    __syncthreads();
  }

  // ---- epilogue: bias + scatter to Q/K/V [B,H,T,Dh]
#pragma unroll
  for (int mf = 0; mf < 4; ++mf) {
#pragma unroll
    for (int nf = 0; nf < 2; ++nf) {
      int j     = n0 + wn * 32 + nf * 16 + l;   // global output column 0..3071
      float bb  = bias[j];
      int which = j >> 10;                       // 0=Q 1=K 2=V
      int rem   = j & 1023;
      int h     = rem >> 6;
      int d     = rem & 63;
      __bf16* dst = (which == 0) ? Qb : ((which == 1) ? Kb : Vb);
      float scale = (which == 0) ? 0.125f : 1.0f;   // 1/sqrt(64)
#pragma unroll
      for (int r = 0; r < 8; ++r) {
        int mg = m0 + wm * 64 + mf * 16 + hi * 8 + r;   // flattened b*T + t
        int b_ = mg >> 11;
        int t  = mg & 2047;
        float v = (acc[mf][nf][r] + bb) * scale;
        dst[(((size_t)(b_ * HEADS + h) * SEQ) + t) * HDIM + d] = (__bf16)v;
      }
    }
  }
}

// ---------------------------------------------------------------------------
// Kernel 2: causal flash attention. One block = (b,h) x 128 query rows.
// 8 waves; each wave owns 16 query rows. K/V streamed in 64-key tiles.
// ---------------------------------------------------------------------------
#define QSTR 72   // padded LDS row stride for 64-wide tiles (144B, 16B-aligned)

__global__ __launch_bounds__(256) void attn_kernel(
    const __bf16* __restrict__ Qb, const __bf16* __restrict__ Kb,
    const __bf16* __restrict__ Vb, __bf16* __restrict__ Ob /* [B,T,C] bf16 */)
{
  __shared__ __bf16 Qs[128 * QSTR];
  __shared__ __bf16 Ks[64 * QSTR];
  __shared__ __bf16 Vt[64 * QSTR];        // transposed: [d][key]
  __shared__ __bf16 Ps[8 * 16 * QSTR];    // per-wave P scratch (16x64 each)

  const int tid  = threadIdx.x;
  const int lane = tid & 31;
  const int wave = tid >> 5;
  const int hi   = lane >> 4;
  const int l    = lane & 15;

  const int bh = blockIdx.y;              // 0..63
  const int q0 = blockIdx.x * 128;
  const __bf16* Qg = Qb + (size_t)bh * SEQ * HDIM;
  const __bf16* Kg = Kb + (size_t)bh * SEQ * HDIM;
  const __bf16* Vg = Vb + (size_t)bh * SEQ * HDIM;

  // ---- Q tile -> LDS (128x64 bf16, 16B chunks)
#pragma unroll
  for (int i = 0; i < 4; ++i) {
    int c   = tid + i * 256;              // chunk of 8 bf16, 0..1023
    int row = c >> 3;
    int c8  = c & 7;
    *(uint4*)&Qs[row * QSTR + c8 * 8] =
        *(const uint4*)&Qg[(size_t)(q0 + row) * HDIM + c8 * 8];
  }
  __syncthreads();

  // wave-resident Q fragments (Dh=64 -> two k-steps of 32)
  v16bf qf[2];
  {
    const __bf16* qb0 = &Qs[(wave * 16 + l) * QSTR];
    qf[0] = frag16(qb0 + hi * 8,       qb0 + 16 + hi * 8);
    qf[1] = frag16(qb0 + 32 + hi * 8,  qb0 + 48 + hi * 8);
  }

  float mi[8], li[8];
  v8f of[4];
#pragma unroll
  for (int r = 0; r < 8; ++r) { mi[r] = -1e30f; li[r] = 0.0f; }
#pragma unroll
  for (int nf = 0; nf < 4; ++nf) of[nf] = zero8();

  const int qrow_base = q0 + wave * 16 + hi * 8;

  for (int kt0 = 0; kt0 < q0 + 128; kt0 += 64) {   // causal: skip tiles above diag
    // ---- K tile 64x64 -> LDS
#pragma unroll
    for (int i = 0; i < 2; ++i) {
      int c   = tid + i * 256;            // 0..511
      int row = c >> 3;
      int c8  = c & 7;
      *(uint4*)&Ks[row * QSTR + c8 * 8] =
          *(const uint4*)&Kg[(size_t)(kt0 + row) * HDIM + c8 * 8];
    }
    // ---- V tile 64x64 -> LDS transposed ([d][key])
    {
      int key   = tid >> 2;
      int dbase = (tid & 3) * 16;
      uint4 a = *(const uint4*)&Vg[(size_t)(kt0 + key) * HDIM + dbase];
      uint4 b = *(const uint4*)&Vg[(size_t)(kt0 + key) * HDIM + dbase + 8];
      const __bf16* pa = (const __bf16*)&a;
      const __bf16* pb = (const __bf16*)&b;
#pragma unroll
      for (int jj = 0; jj < 8; ++jj) {
        Vt[(dbase + jj)     * QSTR + key] = pa[jj];
        Vt[(dbase + 8 + jj) * QSTR + key] = pb[jj];
      }
    }
    __syncthreads();

    // ---- S = Q K^T  (4 n-fragments of 16 keys, 2 k-steps over Dh)
    v8f sf[4];
#pragma unroll
    for (int nf = 0; nf < 4; ++nf) {
      const __bf16* kb = &Ks[(nf * 16 + l) * QSTR];
      v16bf b0 = frag16(kb + hi * 8,      kb + 16 + hi * 8);
      v16bf b1 = frag16(kb + 32 + hi * 8, kb + 48 + hi * 8);
      v8f c = zero8();
      c = WMMA_BF16(qf[0], b0, c);
      c = WMMA_BF16(qf[1], b1, c);
      sf[nf] = c;
    }

    // ---- causal mask (in-register)
#pragma unroll
    for (int nf = 0; nf < 4; ++nf) {
      int kcol = kt0 + nf * 16 + l;
#pragma unroll
      for (int r = 0; r < 8; ++r)
        if (kcol > qrow_base + r) sf[nf][r] = -1e30f;
    }

    // ---- online softmax row statistics (16-lane reductions via ds_bpermute)
    float corr[8];
#pragma unroll
    for (int r = 0; r < 8; ++r) {
      float mx = fmaxf(fmaxf(sf[0][r], sf[1][r]), fmaxf(sf[2][r], sf[3][r]));
#pragma unroll
      for (int msk = 1; msk < 16; msk <<= 1) mx = fmaxf(mx, shfl_xor_f(mx, msk));
      float mnew = fmaxf(mi[r], mx);
      corr[r] = __expf(mi[r] - mnew);
      mi[r]   = mnew;
    }
#pragma unroll
    for (int nf = 0; nf < 4; ++nf)
#pragma unroll
      for (int r = 0; r < 8; ++r) of[nf][r] *= corr[r];

    float rs[8];
#pragma unroll
    for (int r = 0; r < 8; ++r) rs[r] = 0.0f;
#pragma unroll
    for (int nf = 0; nf < 4; ++nf)
#pragma unroll
      for (int r = 0; r < 8; ++r) {
        float p = __expf(sf[nf][r] - mi[r]);
        sf[nf][r] = p;
        rs[r] += p;
      }
#pragma unroll
    for (int r = 0; r < 8; ++r) {
      float s = rs[r];
#pragma unroll
      for (int msk = 1; msk < 16; msk <<= 1) s += shfl_xor_f(s, msk);
      li[r] = li[r] * corr[r] + s;
    }

    // ---- round-trip P through per-wave LDS to re-fragment as WMMA-A
    __bf16* pw = &Ps[wave * 16 * QSTR];
#pragma unroll
    for (int nf = 0; nf < 4; ++nf)
#pragma unroll
      for (int r = 0; r < 8; ++r)
        pw[(hi * 8 + r) * QSTR + nf * 16 + l] = (__bf16)sf[nf][r];

    {
      const __bf16* pb0 = &Ps[(wave * 16 + l) * QSTR];
      v16bf pa0 = frag16(pb0 + hi * 8,      pb0 + 16 + hi * 8);
      v16bf pa1 = frag16(pb0 + 32 + hi * 8, pb0 + 48 + hi * 8);
#pragma unroll
      for (int nf = 0; nf < 4; ++nf) {
        const __bf16* vb = &Vt[(nf * 16 + l) * QSTR];
        v16bf b0 = frag16(vb + hi * 8,      vb + 16 + hi * 8);
        v16bf b1 = frag16(vb + 32 + hi * 8, vb + 48 + hi * 8);
        of[nf] = WMMA_BF16(pa0, b0, of[nf]);
        of[nf] = WMMA_BF16(pa1, b1, of[nf]);
      }
    }
    __syncthreads();
  }

  // ---- normalize + write O to [B,T,C] bf16 (C col = h*64 + d)
  const int b_ = bh >> 4;
  const int h  = bh & 15;
#pragma unroll
  for (int r = 0; r < 8; ++r) {
    float inv = 1.0f / li[r];
    int t = q0 + wave * 16 + hi * 8 + r;
    __bf16* dst = Ob + ((size_t)(b_ * SEQ + t)) * EMBED + h * HDIM;
#pragma unroll
    for (int nf = 0; nf < 4; ++nf)
      dst[nf * 16 + l] = (__bf16)(of[nf][r] * inv);
  }
}

// ---------------------------------------------------------------------------
// Kernel 3: output projection.  out = attn(bf16) @ Wout^T + b_out  (fp32 out)
// ---------------------------------------------------------------------------
__global__ __launch_bounds__(256) void out_gemm_kernel(
    const __bf16* __restrict__ A,    // [8192,1024] bf16
    const float*  __restrict__ W,    // [1024,1024] fp32 row-major
    const float*  __restrict__ bias, // [1024]
    float* __restrict__ Out)         // [8192,1024] fp32
{
  __shared__ __bf16 As[BM * LDA];
  __shared__ __bf16 Bs[BN * LDA];

  const int tid  = threadIdx.x;
  const int lane = tid & 31;
  const int wave = tid >> 5;
  const int wm   = wave >> 2;
  const int wn   = wave & 3;
  const int hi   = lane >> 4;
  const int l    = lane & 15;

  const int m0 = blockIdx.y * BM;
  const int n0 = blockIdx.x * BN;

  v8f acc[4][2];
#pragma unroll
  for (int mf = 0; mf < 4; ++mf)
#pragma unroll
    for (int nf = 0; nf < 2; ++nf) acc[mf][nf] = zero8();

  for (int k0 = 0; k0 < EMBED; k0 += BK) {
    // A tile (bf16 source): 512 chunks of 8 bf16, 2 per thread
#pragma unroll
    for (int i = 0; i < 2; ++i) {
      int c   = tid + i * 256;
      int row = c >> 2;                   // 4 chunks per 32-wide row
      int c8  = c & 3;
      *(uint4*)&As[row * LDA + c8 * 8] =
          *(const uint4*)&A[(size_t)(m0 + row) * EMBED + k0 + c8 * 8];
    }
    // B tile (fp32 -> bf16 via packed cvt)
#pragma unroll
    for (int i = 0; i < 4; ++i) {
      int idx = tid + i * 256;
      int row = idx >> 3;
      int c4  = idx & 7;
      v4f f = *(const v4f*)(W + (size_t)(n0 + row) * EMBED + k0 + c4 * 4);
      *(v4bf*)&Bs[row * LDA + c4 * 4] = cvt4(f);
    }
    if (k0 + BK < EMBED)
      __builtin_prefetch(A + (size_t)(m0 + (tid >> 1)) * EMBED + k0 + BK, 0, 1);
    __syncthreads();

    v16bf bf[2];
#pragma unroll
    for (int nf = 0; nf < 2; ++nf) {
      const __bf16* base = &Bs[(wn * 32 + nf * 16 + l) * LDA];
      bf[nf] = frag16(base + hi * 8, base + 16 + hi * 8);
    }
#pragma unroll
    for (int mf = 0; mf < 4; ++mf) {
      const __bf16* abase = &As[(wm * 64 + mf * 16 + l) * LDA];
      v16bf af = frag16(abase + hi * 8, abase + 16 + hi * 8);
#pragma unroll
      for (int nf = 0; nf < 2; ++nf)
        acc[mf][nf] = WMMA_BF16(af, bf[nf], acc[mf][nf]);
    }
    __syncthreads();
  }

#pragma unroll
  for (int mf = 0; mf < 4; ++mf)
#pragma unroll
    for (int nf = 0; nf < 2; ++nf) {
      int j = n0 + wn * 32 + nf * 16 + l;
      float bb = bias[j];
#pragma unroll
      for (int r = 0; r < 8; ++r) {
        int mg = m0 + wm * 64 + mf * 16 + hi * 8 + r;
        Out[(size_t)mg * EMBED + j] = acc[mf][nf][r] + bb;
      }
    }
}

// ---------------------------------------------------------------------------
// Host launcher
// ---------------------------------------------------------------------------
extern "C" void kernel_launch(void* const* d_in, const int* in_sizes, int n_in,
                              void* d_out, int out_size, void* d_ws, size_t ws_size,
                              hipStream_t stream) {
  (void)in_sizes; (void)n_in; (void)out_size; (void)ws_size;
  const float* x     = (const float*)d_in[0];
  const float* Wqkv  = (const float*)d_in[1];
  const float* bqkv  = (const float*)d_in[2];
  const float* Wout  = (const float*)d_in[3];
  const float* bout  = (const float*)d_in[4];
  // d_in[5] = mask: causality handled analytically in-kernel

  const size_t per = (size_t)BATCH * HEADS * SEQ * HDIM;   // 8,388,608 elems
  __bf16* Qb = (__bf16*)d_ws;
  __bf16* Kb = Qb + per;
  __bf16* Vb = Kb + per;
  __bf16* Ob = Vb + per;   // [B,T,C] bf16  (total ws use: 64 MB)

  qkv_gemm_kernel<<<dim3(3 * EMBED / BN, MTOT / BM), 256, 0, stream>>>(
      x, Wqkv, bqkv, Qb, Kb, Vb);
  attn_kernel<<<dim3(SEQ / 128, BATCH * HEADS), 256, 0, stream>>>(Qb, Kb, Vb, Ob);
  out_gemm_kernel<<<dim3(EMBED / BN, MTOT / BM), 256, 0, stream>>>(
      Ob, Wout, bout, (float*)d_out);
}
